// AttentionTimRNN_59047210385626
// MI455X (gfx1250) — matrix operand order repaired and checked by reading, not compiled, not run on hardware
//
#include <hip/hip_runtime.h>
#include <hip/hip_bf16.h>

// Problem constants (from reference): T=256, B=128, H=512, E=256, IN=512, D=1024
static constexpr int cT  = 256;
static constexpr int cB  = 128;
static constexpr int cH  = 512;
static constexpr int cE  = 256;
static constexpr int cIN = 512;   // 2*E
static constexpr int cD  = 1024;  // 2*H
static constexpr int c3H = 1536;

typedef __bf16 bf16x16 __attribute__((ext_vector_type(16)));
typedef __bf16 bf16x8  __attribute__((ext_vector_type(8)));
typedef float  f32x8   __attribute__((ext_vector_type(8)));
typedef unsigned int u32x4 __attribute__((ext_vector_type(4)));
typedef int i32x4 __attribute__((ext_vector_type(4)));
typedef int i32x8 __attribute__((ext_vector_type(8)));

union BFrag { bf16x16 v; bf16x8 h[2]; };

__device__ __forceinline__ __bf16 f2bf(float x) { return (__bf16)x; }

#if defined(__HIP_DEVICE_COMPILE__) && __has_builtin(__builtin_amdgcn_tensor_load_to_lds) && __has_builtin(__builtin_amdgcn_s_wait_tensorcnt)
#define USE_TDM 1
#else
#define USE_TDM 0
#endif

#if USE_TDM
// Issue one TDM 2D tile load: tile_d1 rows x tile_d0 elements of 2-byte data,
// row stride = stride0 elements, global start = gaddr, into LDS at lds_addr.
// D# packing per CDNA5 ISA (group0: count=1,
// lds_addr, 57-bit global addr, type=2; group1: data_size=2B, dims/tile/stride).
// This toolchain (clang-23 / therock-10.0 headers) exposes the 6-arg builtin:
// (u32x4 g0, i32x8 g1, i32x4 g2, i32x4 g3, i32x8 extra, i32 cpol).
__device__ __forceinline__ void tdm_load_2d(unsigned lds_addr, const void* gaddr,
                                            unsigned tensor_d0, unsigned tensor_d1,
                                            unsigned tile_d0, unsigned tile_d1,
                                            unsigned long long stride0) {
    unsigned long long ga = (unsigned long long)(size_t)gaddr;
    u32x4 g0 = { 1u,                                   // count=1, user mode
                 lds_addr,                             // LDS byte offset
                 (unsigned)ga,                         // global_addr[31:0]
                 (unsigned)((ga >> 32) & 0x01FFFFFFu)  // global_addr[56:32]
                     | (2u << 30) };                   // type = 2 ("image")
    i32x8 g1 = { (int)0x00010000u,                             // data_size=1 (2 bytes)
                 (int)((tensor_d0 & 0xFFFFu) << 16),           // tensor_dim0[15:0]
                 (int)((tensor_d0 >> 16) | ((tensor_d1 & 0xFFFFu) << 16)),
                 (int)((tensor_d1 >> 16) | (tile_d0 << 16)),   // tile_dim0
                 (int)(tile_d1),                               // tile_dim1 (tile_dim2=0)
                 (int)(unsigned)(stride0 & 0xFFFFFFFFu),       // dim0_stride[31:0]
                 (int)(unsigned)((stride0 >> 32) & 0xFFFFu),   // dim0_stride[47:32]
                 0 };
    i32x4 z4 = { 0, 0, 0, 0 };
    i32x8 z8 = { 0, 0, 0, 0, 0, 0, 0, 0 };
    __builtin_amdgcn_tensor_load_to_lds(g0, g1, z4, z4, z8, 0);
}
#endif

// ---------------------------------------------------------------------------
// bf16 GEMM, fp32 accumulate:  C[m,n] = sum_k A[m,k]*Bt[n,k] + bias[n]
// A: (M,K) bf16 row-major; Bt: (N,K) bf16 row-major ("W" as stored, for x@W.T)
// Block tile 128x128, K-step 32, 8 waves; each wave: 16(M) x 128(N) = 8 WMMA
// accumulators. TDM double-buffered global->LDS staging (wave 0 drives DMA).
// Fragment layouts (ISA 7.12.2, wave32):
//   A 16x32: lane<16 -> row=lane, K={0..7}|{16..23}; lane>=16 -> K={8..15}|{24..31}
//   B 32x16: lane<16 -> col=lane, K={0..7}|{16..23}; lane>=16 -> K={8..15}|{24..31}
// Both are contiguous 16B chunks in row-major (row, K) LDS tiles.
// ---------------------------------------------------------------------------
__global__ __launch_bounds__(256)
void gemm_bf16_wmma(const __bf16* __restrict__ A, const __bf16* __restrict__ Bt,
                    const float* __restrict__ bias, float* __restrict__ C,
                    int M, int N, int K) {
    constexpr int BM = 128, BN = 128, BK = 32;
    __shared__ __align__(16) __bf16 sA[2][BM * BK];
    __shared__ __align__(16) __bf16 sB[2][BN * BK];

    const int tid  = threadIdx.x;
    const int m0   = blockIdx.y * BM;
    const int n0   = blockIdx.x * BN;
    const int wv   = tid >> 5;
    const int lane = tid & 31;
    const int l16  = lane & 15;
    const int hi   = lane >> 4;

    f32x8 acc[8] = {};
    const int nt = K / BK;

    auto compute = [&](int cur) {
        BFrag af;
        const __bf16* ap = &sA[cur][(wv * 16 + l16) * BK + hi * 8];
        af.h[0] = *(const bf16x8*)(ap);
        af.h[1] = *(const bf16x8*)(ap + 16);
#pragma unroll
        for (int j = 0; j < 8; ++j) {
            BFrag bfr;
            const __bf16* bp = &sB[cur][(j * 16 + l16) * BK + hi * 8];
            bfr.h[0] = *(const bf16x8*)(bp);
            bfr.h[1] = *(const bf16x8*)(bp + 16);
            acc[j] = __builtin_amdgcn_wmma_f32_16x16x32_bf16(
                false, af.v, false, bfr.v, (short)0, acc[j], false, false);
        }
    };

#if USE_TDM
    const __bf16* Abase = A + (size_t)m0 * K;
    const __bf16* Bbase = Bt + (size_t)n0 * K;
    if (wv == 0) {
        tdm_load_2d((unsigned)(size_t)&sA[0][0], Abase, (unsigned)K, (unsigned)M,
                    BK, BM, (unsigned long long)K);
        tdm_load_2d((unsigned)(size_t)&sB[0][0], Bbase, (unsigned)K, (unsigned)N,
                    BK, BN, (unsigned long long)K);
    }
    for (int i = 0; i < nt; ++i) {
        const int cur = i & 1;
        const int nxt = cur ^ 1;
        if (wv == 0) {
            if (i + 1 < nt) {
                tdm_load_2d((unsigned)(size_t)&sA[nxt][0], Abase + (size_t)(i + 1) * BK,
                            (unsigned)K, (unsigned)M, BK, BM, (unsigned long long)K);
                tdm_load_2d((unsigned)(size_t)&sB[nxt][0], Bbase + (size_t)(i + 1) * BK,
                            (unsigned)K, (unsigned)N, BK, BN, (unsigned long long)K);
                __builtin_amdgcn_s_wait_tensorcnt(2);  // tile i's two DMAs done
            } else {
                __builtin_amdgcn_s_wait_tensorcnt(0);
            }
        }
        __syncthreads();   // tile i resident for all waves
        compute(cur);
        __syncthreads();   // all reads done before wave 0 overwrites this buffer
    }
#else
    // Fallback: straight-line synchronous staging (fixed chunk per thread).
    const int r = tid >> 2;
    const int c = (tid & 3) << 3;
    const __bf16* Ar0 = A + (size_t)(m0 + r) * K + c;
    const __bf16* Ar1 = A + (size_t)(m0 + r + 64) * K + c;
    const __bf16* Br0 = Bt + (size_t)(n0 + r) * K + c;
    const __bf16* Br1 = Bt + (size_t)(n0 + r + 64) * K + c;
    for (int i = 0; i < nt; ++i) {
        const size_t k0 = (size_t)i * BK;
        *(uint4*)&sA[0][r * BK + c]        = *(const uint4*)(Ar0 + k0);
        *(uint4*)&sA[0][(r + 64) * BK + c] = *(const uint4*)(Ar1 + k0);
        *(uint4*)&sB[0][r * BK + c]        = *(const uint4*)(Br0 + k0);
        *(uint4*)&sB[0][(r + 64) * BK + c] = *(const uint4*)(Br1 + k0);
        __syncthreads();
        compute(0);
        __syncthreads();
    }
#endif

    // C layout: VGPR r, lanes 0-15 -> M=r, N=lane; lanes 16-31 -> M=r+8.
#pragma unroll
    for (int j = 0; j < 8; ++j) {
        int n = n0 + j * 16 + l16;
        float bv = bias[n];
#pragma unroll
        for (int r2 = 0; r2 < 8; ++r2) {
            int m = m0 + wv * 16 + hi * 8 + r2;
            C[(size_t)m * N + n] = acc[j][r2] + bv;
        }
    }
}

// ---------------------------------------------------------------------------
// Small helper kernels
// ---------------------------------------------------------------------------
__global__ void cvt_bf16(const float* __restrict__ src, __bf16* __restrict__ dst, int n) {
    int i = blockIdx.x * 256 + threadIdx.x;
    if (i < n) dst[i] = f2bf(src[i]);
}

// dst[c*R + r] = src[r*C + c]  (gives Bt[n,k] = W_att[k,n])
__global__ void transpose_bf16(const float* __restrict__ src, __bf16* __restrict__ dst,
                               int R, int C) {
    int i = blockIdx.x * 256 + threadIdx.x;
    if (i < R * C) {
        int r = i / C, c = i % C;
        dst[(size_t)c * R + r] = f2bf(src[i]);
    }
}

// embedded[m, 0:256] = emb_tim[tim[m]], embedded[m, 256:512] = emb_loc[loc[m]]
__global__ void embed_kernel(const int* __restrict__ tim, const int* __restrict__ loc,
                             const float* __restrict__ Et, const float* __restrict__ El,
                             __bf16* __restrict__ out) {
    int m = blockIdx.x;
    int j = threadIdx.x;  // 256 threads == E
    int ti = tim[m], li = loc[m];
    out[(size_t)m * cIN + j]      = f2bf(Et[(size_t)ti * cE + j]);
    out[(size_t)m * cIN + cE + j] = f2bf(El[(size_t)li * cE + j]);
}

__global__ void init_h(const float* __restrict__ state,
                       float* __restrict__ h32f, float* __restrict__ h32b,
                       __bf16* __restrict__ hbff, __bf16* __restrict__ hbfb) {
    int i = blockIdx.x * 256 + threadIdx.x;  // B*H
    float f = state[i];
    float b = state[cB * cH + i];
    h32f[i] = f; hbff[i] = f2bf(f);
    h32b[i] = b; hbfb[i] = f2bf(b);
}

// One GRU step elementwise (B*H threads).
__global__ void gru_gates(const float* __restrict__ Gt, const float* __restrict__ GH,
                          float* __restrict__ h32, __bf16* __restrict__ hbf,
                          float* __restrict__ outF, __bf16* __restrict__ outB) {
    int i = blockIdx.x * 256 + threadIdx.x;
    int b = i >> 9;        // /H
    int j = i & (cH - 1);  // %H
    size_t g = (size_t)b * c3H;
    float r = 1.f / (1.f + expf(-(Gt[g + j]      + GH[g + j])));
    float z = 1.f / (1.f + expf(-(Gt[g + cH + j] + GH[g + cH + j])));
    float n = tanhf(Gt[g + 2 * cH + j] + r * GH[g + 2 * cH + j]);
    float hp = h32[i];
    float hn = (1.f - z) * n + z * hp;
    h32[i] = hn;
    hbf[i] = f2bf(hn);
    size_t o = (size_t)b * cD + j;
    outF[o] = hn;
    outB[o] = f2bf(hn);
}

// scores[m] = sum_n tanh(U[m,n]) * w_proj[n]   (U already includes b_att)
__global__ void scores_kernel(const float* __restrict__ U,
                              const float* __restrict__ w_proj,
                              float* __restrict__ scores) {
    __shared__ float red[256];
    int m = blockIdx.x;
    int tid = threadIdx.x;
    float s = 0.f;
    for (int n = tid; n < cD; n += 256)
        s += tanhf(U[(size_t)m * cD + n]) * w_proj[n];
    red[tid] = s;
    __syncthreads();
    for (int off = 128; off > 0; off >>= 1) {
        if (tid < off) red[tid] += red[tid + off];
        __syncthreads();
    }
    if (tid == 0) scores[m] = red[0];
}

// scores (T,B) -> attn (B,T), softmax over T per batch
__global__ void softmax_kernel(const float* __restrict__ scores, float* __restrict__ attn) {
    __shared__ float sh[cT];
    int b = blockIdx.x;
    int t = threadIdx.x;  // 256 == T
    float v = scores[(size_t)t * cB + b];
    sh[t] = v;
    __syncthreads();
    for (int off = 128; off > 0; off >>= 1) {
        if (t < off) sh[t] = fmaxf(sh[t], sh[t + off]);
        __syncthreads();
    }
    float mx = sh[0];
    __syncthreads();
    float e = expf(v - mx);
    sh[t] = e;
    __syncthreads();
    for (int off = 128; off > 0; off >>= 1) {
        if (t < off) sh[t] += sh[t + off];
        __syncthreads();
    }
    attn[(size_t)b * cT + t] = e / sh[0];
}

// pooled[b,d] = sum_t attn[b,t] * out[t,b,d]
__global__ void pooled_kernel(const float* __restrict__ attn, const float* __restrict__ out,
                              float* __restrict__ pooled) {
    __shared__ float a[cT];
    int b = blockIdx.x;
    int tid = threadIdx.x;
    a[tid] = attn[(size_t)b * cT + tid];
    __syncthreads();
    for (int d = tid; d < cD; d += 256) {
        float acc = 0.f;
        for (int t = 0; t < cT; ++t)
            acc += a[t] * out[((size_t)t * cB + b) * cD + d];
        pooled[(size_t)b * cD + d] = acc;
    }
}

// ---------------------------------------------------------------------------
extern "C" void kernel_launch(void* const* d_in, const int* in_sizes, int n_in,
                              void* d_out, int out_size, void* d_ws, size_t ws_size,
                              hipStream_t stream) {
    const int*   tim_seq   = (const int*)d_in[0];
    const int*   loc_seq   = (const int*)d_in[1];
    const float* state_tim = (const float*)d_in[2];
    const float* emb_tim   = (const float*)d_in[3];
    const float* emb_loc   = (const float*)d_in[4];
    const float* Wih_f     = (const float*)d_in[5];
    const float* Whh_f     = (const float*)d_in[6];
    const float* bih_f     = (const float*)d_in[7];
    const float* bhh_f     = (const float*)d_in[8];
    const float* Wih_b     = (const float*)d_in[9];
    const float* Whh_b     = (const float*)d_in[10];
    const float* bih_b     = (const float*)d_in[11];
    const float* bhh_b     = (const float*)d_in[12];
    const float* W_att     = (const float*)d_in[13];
    const float* b_att     = (const float*)d_in[14];
    const float* w_proj    = (const float*)d_in[15];

    char*  ws  = (char*)d_ws;
    size_t off = 0;
    auto take = [&](size_t bytes) -> char* {
        char* p = ws + off;
        off = (off + bytes + 255) & ~(size_t)255;
        return p;
    };
    const size_t MTB = (size_t)cT * cB;  // 32768

    __bf16* embCat = (__bf16*)take(MTB * cIN * 2);
    __bf16* wihf   = (__bf16*)take((size_t)c3H * cIN * 2);
    __bf16* whhf   = (__bf16*)take((size_t)c3H * cH * 2);
    __bf16* wihb   = (__bf16*)take((size_t)c3H * cIN * 2);
    __bf16* whhb   = (__bf16*)take((size_t)c3H * cH * 2);
    __bf16* wattT  = (__bf16*)take((size_t)cD * cD * 2);
    float*  Gf     = (float*)take(MTB * c3H * 4);   // reused as U after the RNN loop
    float*  Gb     = (float*)take(MTB * c3H * 4);
    float*  GHf    = (float*)take((size_t)cB * c3H * 4);
    float*  GHb    = (float*)take((size_t)cB * c3H * 4);
    float*  h32f   = (float*)take((size_t)cB * cH * 4);
    float*  h32b   = (float*)take((size_t)cB * cH * 4);
    __bf16* hbff   = (__bf16*)take((size_t)cB * cH * 2);
    __bf16* hbfb   = (__bf16*)take((size_t)cB * cH * 2);
    float*  outF   = (float*)take(MTB * cD * 4);
    __bf16* outBf  = (__bf16*)take(MTB * cD * 2);
    float*  scores = (float*)take(MTB * 4);
    if (off > ws_size) return;  // workspace too small: deterministic no-op

    float* pooledOut = (float*)d_out;                  // (1,B,D)
    float* stateOut  = pooledOut + (size_t)cB * cD;    // (2,B,H)
    float* attnOut   = stateOut + 2 * (size_t)cB * cH; // (B,T)

    // 1) Pack weights to bf16
    cvt_bf16<<<(c3H * cIN + 255) / 256, 256, 0, stream>>>(Wih_f, wihf, c3H * cIN);
    cvt_bf16<<<(c3H * cH + 255) / 256, 256, 0, stream>>>(Whh_f, whhf, c3H * cH);
    cvt_bf16<<<(c3H * cIN + 255) / 256, 256, 0, stream>>>(Wih_b, wihb, c3H * cIN);
    cvt_bf16<<<(c3H * cH + 255) / 256, 256, 0, stream>>>(Whh_b, whhb, c3H * cH);
    transpose_bf16<<<(cD * cD + 255) / 256, 256, 0, stream>>>(W_att, wattT, cD, cD);

    // 2) Embedding gather -> bf16 (T*B, IN)
    embed_kernel<<<(int)MTB, 256, 0, stream>>>(tim_seq, loc_seq, emb_tim, emb_loc, embCat);

    // 3) Batched input-side GEMMs: G = embedded @ Wih.T + bih
    dim3 gG(c3H / 128, (int)(MTB / 128));
    gemm_bf16_wmma<<<gG, 256, 0, stream>>>(embCat, wihf, bih_f, Gf, (int)MTB, c3H, cIN);
    gemm_bf16_wmma<<<gG, 256, 0, stream>>>(embCat, wihb, bih_b, Gb, (int)MTB, c3H, cIN);

    // 4) Init hidden states
    init_h<<<(cB * cH) / 256, 256, 0, stream>>>(state_tim, h32f, h32b, hbff, hbfb);

    // 5) Sequential recurrence
    dim3 gGH(c3H / 128, cB / 128);
    for (int s = 0; s < cT; ++s) {
        int tf = s;
        int tb = cT - 1 - s;
        gemm_bf16_wmma<<<gGH, 256, 0, stream>>>(hbff, whhf, bhh_f, GHf, cB, c3H, cH);
        gru_gates<<<(cB * cH) / 256, 256, 0, stream>>>(
            Gf + (size_t)tf * cB * c3H, GHf, h32f, hbff,
            outF + (size_t)tf * cB * cD, outBf + (size_t)tf * cB * cD);
        gemm_bf16_wmma<<<gGH, 256, 0, stream>>>(hbfb, whhb, bhh_b, GHb, cB, c3H, cH);
        gru_gates<<<(cB * cH) / 256, 256, 0, stream>>>(
            Gb + (size_t)tb * cB * c3H, GHb, h32b, hbfb,
            outF + (size_t)tb * cB * cD + cH, outBf + (size_t)tb * cB * cD + cH);
    }

    // 6) Attention
    float* U = Gf;
    dim3 gU(cD / 128, (int)(MTB / 128));
    gemm_bf16_wmma<<<gU, 256, 0, stream>>>(outBf, wattT, b_att, U, (int)MTB, cD, cD);
    scores_kernel<<<(int)MTB, 256, 0, stream>>>(U, w_proj, scores);
    softmax_kernel<<<cB, cT, 0, stream>>>(scores, attnOut);
    pooled_kernel<<<cB, 256, 0, stream>>>(attnOut, outF, pooledOut);

    // 7) Final states -> d_out (pooled | state | attn)
    (void)hipMemcpyAsync(stateOut, h32f, (size_t)cB * cH * sizeof(float),
                         hipMemcpyDeviceToDevice, stream);
    (void)hipMemcpyAsync(stateOut + (size_t)cB * cH, h32b, (size_t)cB * cH * sizeof(float),
                         hipMemcpyDeviceToDevice, stream);
}
